// KANLayer_46059229282687
// MI455X (gfx1250) — matrix operand order repaired and checked by reading, not compiled, not run on hardware
//
#include <hip/hip_runtime.h>
#include <stdint.h>

typedef __attribute__((ext_vector_type(16))) _Float16 v16h;
typedef __attribute__((ext_vector_type(8)))  float    v8f;

#define B_  256
#define I_  512
#define O_  512
#define G_  20
#define K_  (I_ * G_)        // 10240
#define KT_ (K_ / 32)        // 320 k-tiles of 32
#define KSTEP 4              // k-tiles per pipeline stage
#define NSTAGE (KT_ / KSTEP) // 80 stages

// Workspace layout (bytes). Total = 16 MiB exactly.
#define OFF_IDX   0u                                  // int   [B_*I_]
#define OFF_T     (B_ * I_ * 4u)                      // float [B_*I_]
#define OFF_AFRAG (OFF_T + B_ * I_ * 4u)              // f16   [16*KT_*512]
#define OFF_BFRAG (OFF_AFRAG + 16u * KT_ * 512u * 2u) // f16   [32*KT_*512]

// ---------------------------------------------------------------------------
// CDNA5 async global->LDS copy + wait (ISA 10.x / 15.18, ASYNCcnt tracked)
// ---------------------------------------------------------------------------
__device__ __forceinline__ void async_ld_b128(unsigned lds_addr, const void* gaddr) {
    asm volatile("global_load_async_to_lds_b128 %0, %1, off"
                 :: "v"(lds_addr), "v"(gaddr) : "memory");
}
__device__ __forceinline__ void wait_async0() {
    asm volatile("s_wait_asynccnt 0" ::: "memory");
}

// ---------------------------------------------------------------------------
// Phase 0: per-(b,i) cell index + fraction, exact searchsorted semantics.
// ---------------------------------------------------------------------------
__global__ __launch_bounds__(256) void kan_basis(const float* __restrict__ x,
                                                 const float* __restrict__ grid,
                                                 int* __restrict__ idxOut,
                                                 float* __restrict__ tOut) {
    const int id = blockIdx.x * 256 + threadIdx.x;
    const float g0 = grid[0];
    const float gN = grid[G_ - 1];
    const float xc = fminf(fmaxf(x[id], g0), gN);
    int cnt = 0;
#pragma unroll
    for (int j = 1; j < G_; ++j) cnt += (grid[j] < xc) ? 1 : 0;
    const int idx = cnt > (G_ - 2) ? (G_ - 2) : cnt;
    const float x0 = grid[idx];
    const float x1 = grid[idx + 1];
    idxOut[id] = idx;
    tOut[id]   = (xc - x0) / (x1 - x0 + 1e-8f);
}

// ---------------------------------------------------------------------------
// Phase 1: A fragments (basis matrix), pre-swizzled wave32 16-bit A layout:
//   lane L: M=L%16, lanehi=L/16; element e: K = e + (e>=8?8:0) + lanehi*8
// ---------------------------------------------------------------------------
__global__ __launch_bounds__(256) void kan_afrag(const int* __restrict__ idxIn,
                                                 const float* __restrict__ tIn,
                                                 _Float16* __restrict__ aFrag) {
    const int tid    = blockIdx.x * 256 + threadIdx.x;
    const int lane   = tid & 31;
    const int frag   = tid >> 5;            // frag = mt*KT_ + kt
    const int kt     = frag % KT_;
    const int mt     = frag / KT_;
    const int b      = mt * 16 + (lane & 15);
    const int lanehi = lane >> 4;

    v16h vals;
#pragma unroll
    for (int e = 0; e < 16; ++e) {
        const int k  = kt * 32 + e + ((e >= 8) ? 8 : 0) + lanehi * 8;
        const int i  = k / G_;
        const int g  = k % G_;
        const int ix = idxIn[b * I_ + i];
        const float tv = tIn[b * I_ + i];
        const float c  = (g == ix) ? (1.0f - tv) : ((g == ix + 1) ? tv : 0.0f);
        vals[e] = (_Float16)c;
    }
    *(v16h*)(aFrag + (size_t)frag * 512 + lane * 16) = vals;
}

// ---------------------------------------------------------------------------
// Phase 2: B fragments: W[i*G+g, o] = sw[i,o,g]*scale[i,o] (f16), wave32 B
// layout: lane L: N=L%16, lanehi=L/16; element e: K = e + (e>=8?8:0) + lanehi*8
// ---------------------------------------------------------------------------
__global__ __launch_bounds__(256) void kan_bfrag(const float* __restrict__ sw,
                                                 const float* __restrict__ scale,
                                                 _Float16* __restrict__ bFrag) {
    const int tid    = blockIdx.x * 256 + threadIdx.x;
    const int lane   = tid & 31;
    const int frag   = tid >> 5;            // frag = nt*KT_ + kt
    const int kt     = frag % KT_;
    const int nt     = frag / KT_;
    const int o      = nt * 16 + (lane & 15);
    const int lanehi = lane >> 4;

    v16h vals;
#pragma unroll
    for (int e = 0; e < 16; ++e) {
        const int k = kt * 32 + e + ((e >= 8) ? 8 : 0) + lanehi * 8;
        const int i = k / G_;
        const int g = k % G_;
        const float w = sw[((size_t)i * O_ + o) * G_ + g] * scale[i * O_ + o];
        vals[e] = (_Float16)w;
    }
    *(v16h*)(bFrag + (size_t)frag * 512 + lane * 16) = vals;
}

// ---------------------------------------------------------------------------
// Phase 3: out = C[256,10240] @ W[10240,512], v_wmma_f32_16x16x32_f16.
// Block = 8 waves sharing one N-group (4 N-tiles); each wave owns one M-tile.
// B staged in LDS by async global->LDS copies, double buffered; A register
// double-buffered one stage ahead. All control flow is wave-uniform (EXEC=~0).
// ---------------------------------------------------------------------------
__global__ __launch_bounds__(256) void kan_gemm(const _Float16* __restrict__ aFrag,
                                                const _Float16* __restrict__ bFrag,
                                                float* __restrict__ out) {
    __shared__ __align__(32) _Float16 bsh[2][KSTEP * 4 * 512];   // 2 x 16 KiB

    const int t    = threadIdx.x;
    const int lane = t & 31;
    const int w    = t >> 5;                 // wave in block, 0..7
    const int ng   = blockIdx.x & 7;         // N-group (4 tiles = 64 cols)
    const int mt   = ((blockIdx.x >> 3) * 8) + w;   // M-tile, 0..15

    // ---- per-thread async-copy descriptors (4 x b128 per stage = 16 KiB/block)
    const unsigned ldsBase = (unsigned)(uintptr_t)&bsh[0][0];
    const char* gB[4];
    unsigned    ldsOff[4];
#pragma unroll
    for (int c = 0; c < 4; ++c) {
        const int ch = c * 256 + t;          // chunk 0..1023 of 16B
        const int f  = ch >> 6;              // fragment slot 0..15 = j*KSTEP+s
        const int j  = f >> 2;               // N-tile within group
        const int s  = f & 3;                // k-tile within stage
        const int co = (ch & 63) << 4;       // byte offset inside fragment
        gB[c]     = (const char*)bFrag + (((size_t)(ng * 4 + j) * KT_ + s) << 10) + co;
        ldsOff[c] = ldsBase + (unsigned)(f << 10) + (unsigned)co;
    }

    const v16h* __restrict__ ap = (const v16h*)(aFrag + (size_t)mt * KT_ * 512 + lane * 16);

    v8f  acc[4] = {{}, {}, {}, {}};
    v16h aC[4], aN[4];

    // ---- prologue: stage 0 B -> buf0, A tiles 0..3 -> regs
#pragma unroll
    for (int c = 0; c < 4; ++c) async_ld_b128(ldsOff[c], gB[c]);
#pragma unroll
    for (int s = 0; s < KSTEP; ++s) aC[s] = ap[s * 32];
    wait_async0();
    __syncthreads();

    // ---- main pipeline: issue stage st+1 while computing stage st
    for (int st = 0; st < NSTAGE - 1; ++st) {
        const int nxt = st + 1;
#pragma unroll
        for (int c = 0; c < 4; ++c)
            async_ld_b128(ldsOff[c] + (unsigned)((nxt & 1) * (KSTEP * 4 * 1024)),
                          gB[c] + (size_t)nxt * (KSTEP * 1024));
#pragma unroll
        for (int s = 0; s < KSTEP; ++s) aN[s] = ap[(nxt * KSTEP + s) * 32];

        const _Float16* bb = &bsh[st & 1][0];
#pragma unroll
        for (int s = 0; s < KSTEP; ++s) {
#pragma unroll
            for (int j = 0; j < 4; ++j) {
                const v16h b = *(const v16h*)(bb + (size_t)(j * KSTEP + s) * 512 + lane * 16);
                acc[j] = __builtin_amdgcn_wmma_f32_16x16x32_f16(
                    false, aC[s], false, b, (short)0, acc[j], false, false);
            }
        }
#pragma unroll
        for (int s = 0; s < KSTEP; ++s) aC[s] = aN[s];

        wait_async0();
        __syncthreads();
    }

    // ---- epilogue: last stage
    {
        const _Float16* bb = &bsh[(NSTAGE - 1) & 1][0];
#pragma unroll
        for (int s = 0; s < KSTEP; ++s) {
#pragma unroll
            for (int j = 0; j < 4; ++j) {
                const v16h b = *(const v16h*)(bb + (size_t)(j * KSTEP + s) * 512 + lane * 16);
                acc[j] = __builtin_amdgcn_wmma_f32_16x16x32_f16(
                    false, aC[s], false, b, (short)0, acc[j], false, false);
            }
        }
    }

    // C/D layout: lane holds col N = lane%16; rows M = (lane/16)*8 + v
    const int row = mt * 16 + (lane >> 4) * 8;
    const int col = lane & 15;
#pragma unroll
    for (int v = 0; v < 8; ++v) {
#pragma unroll
        for (int j = 0; j < 4; ++j)
            out[(size_t)(row + v) * O_ + (ng * 4 + j) * 16 + col] = acc[j][v];
    }
}

// ---------------------------------------------------------------------------
extern "C" void kernel_launch(void* const* d_in, const int* in_sizes, int n_in,
                              void* d_out, int out_size, void* d_ws, size_t ws_size,
                              hipStream_t stream) {
    const float* x     = (const float*)d_in[0];   // [B, I]
    const float* grid  = (const float*)d_in[1];   // [G]
    const float* sw    = (const float*)d_in[2];   // [I, O, G]
    const float* scale = (const float*)d_in[3];   // [I, O]
    float*       out   = (float*)d_out;           // [B, O]

    char* ws = (char*)d_ws;                       // needs 16 MiB
    int*      idxBuf = (int*)(ws + OFF_IDX);
    float*    tBuf   = (float*)(ws + OFF_T);
    _Float16* aFrag  = (_Float16*)(ws + OFF_AFRAG);
    _Float16* bFrag  = (_Float16*)(ws + OFF_BFRAG);

    kan_basis<<<(B_ * I_) / 256, 256, 0, stream>>>(x, grid, idxBuf, tBuf);
    kan_afrag<<<(16 * KT_ * 32) / 256, 256, 0, stream>>>(idxBuf, tBuf, aFrag);
    kan_bfrag<<<(32 * KT_ * 32) / 256, 256, 0, stream>>>(sw, scale, bFrag);
    kan_gemm<<<16, 256, 0, stream>>>(aFrag, bFrag, out);
}
